// SarvamMoEMLAAttention_3341484557059
// MI455X (gfx1250) — matrix operand order repaired and checked by reading, not compiled, not run on hardware
//
#include <hip/hip_runtime.h>
#include <hip/hip_bf16.h>
#include <stdint.h>

#define T_TOK   2048
#define HID     4096
#define H       32
#define NOPE    128
#define ROPE    64
#define VD      128
#define KV_RANK 512
#define Q_RANK  1536
#define DQK     192        // NOPE + ROPE
#define EPS     1e-6f
#define SCALE   0.07216878364870323f   // 192^-0.5
#define LN10000 9.210340371976184f

typedef __attribute__((ext_vector_type(16))) __bf16 v16bf;
typedef __attribute__((ext_vector_type(8)))  float  v8f;

union Frag16 { v16bf v; unsigned short h[16]; unsigned int u[8]; };

__device__ __forceinline__ unsigned short f2bf(float f) {
  unsigned int u = __float_as_uint(f);
  unsigned int r = u + 0x7FFFu + ((u >> 16) & 1u);   // round-to-nearest-even
  return (unsigned short)(r >> 16);
}

// A-fragment (16-bit, 16x32) K index of dword v (0..7) for lane-half (ISA 7.12.2):
// VGPR0..3: K = 2v + 8*half ; VGPR4..7: K = 16 + 2(v-4) + 8*half
__device__ __forceinline__ int kidxA(int v, int half) {
  return 2 * v + 8 * half + ((v >= 4) ? 8 : 0);
}

// CDNA5 async memory->LDS copy (ASYNCcnt-tracked, no VGPR round trip).
// LDS destination = low 32 bits of the generic pointer (aperture rule).
__device__ __forceinline__ void async_copy_b128(const void* gptr, void* lptr) {
  asm volatile("global_load_async_to_lds_b128 %0, %1, off"
               :: "v"((unsigned)(uintptr_t)lptr),
                  "v"((unsigned long long)(uintptr_t)gptr)
               : "memory");
}
__device__ __forceinline__ void async_wait0() {
  asm volatile("s_wait_asynccnt 0x0" ::: "memory");
}

// ---------------------------------------------------------------- convert
__global__ void k_cvt_bf16(const float* __restrict__ in,
                           unsigned short* __restrict__ out, int n) {
  int i = blockIdx.x * blockDim.x + threadIdx.x;
  int stride = gridDim.x * blockDim.x;
  for (; i < n; i += stride) out[i] = f2bf(in[i]);
}

// ---------------------------------------------------------------- GEMM
// C[M,N](f32) = A[M,K](bf16 rm) @ B[K,N](bf16 rm).  M%128==0, N%64==0, K%32==0.
// Block tile 128x64, 8 waves, each wave owns a 32x32 C sub-tile (4 WMMA/chunk).
// Double-buffered LDS: async DMA of tile i+1 overlaps WMMA burst on tile i.
__global__ __launch_bounds__(256) void k_gemm_bf16(
    const unsigned short* __restrict__ A, const unsigned short* __restrict__ B,
    float* __restrict__ C, int M, int N, int K) {
  __shared__ unsigned short As[2][128][32];
  __shared__ unsigned short BsT[2][64][32];   // [n][k] transposed for aligned gathers
  const int tid  = threadIdx.x;
  const int wave = tid >> 5;
  const int lane = tid & 31;
  const int half = lane >> 4;
  const int ln16 = lane & 15;
  const int wm = wave & 3;    // 4 row tiles of 32
  const int wn = wave >> 2;   // 2 col tiles of 32
  const int m0 = blockIdx.y * 128;
  const int n0 = blockIdx.x * 64;

  v8f acc[2][2] = {};

  const int ar = tid >> 2;        // A tile: 2 async b128 per thread (128x32)
  const int ac = (tid & 3) * 8;
  const int br = tid >> 3;        // B tile: 8 ushorts of one row (32x64)
  const int bc = (tid & 7) * 8;

  auto stage = [&](int buf, int k0) {
#pragma unroll
    for (int j = 0; j < 2; ++j)
      async_copy_b128(&A[(size_t)(m0 + ar + 64 * j) * K + k0 + ac],
                      &As[buf][ar + 64 * j][ac]);
    uint4 t = *(const uint4*)&B[(size_t)(k0 + br) * N + n0 + bc];
    const unsigned short* pb = (const unsigned short*)&t;
#pragma unroll
    for (int j = 0; j < 8; ++j) BsT[buf][bc + j][br] = pb[j];
  };

  stage(0, 0);
  int buf = 0;
  for (int k0 = 0; k0 < K; k0 += 32, buf ^= 1) {
    async_wait0();
    __syncthreads();
    if (k0 + 32 < K) {
      stage(buf ^ 1, k0 + 32);
      if (k0 + 64 < K) {
        __builtin_prefetch(&A[(size_t)(m0 + ar) * K + k0 + 64 + ac], 0, 0);
        __builtin_prefetch(&B[(size_t)(k0 + 64 + br) * N + n0 + bc], 0, 0);
      }
    }

    Frag16 a[2], b[2];
#pragma unroll
    for (int mt = 0; mt < 2; ++mt)
#pragma unroll
      for (int vI = 0; vI < 8; ++vI)
        a[mt].u[vI] =
            *(const unsigned int*)&As[buf][wm * 32 + mt * 16 + ln16][2 * vI + 16 * half];
#pragma unroll
    for (int nt = 0; nt < 2; ++nt)
#pragma unroll
      for (int vI = 0; vI < 8; ++vI)
        b[nt].u[vI] =
            *(const unsigned int*)&BsT[buf][wn * 32 + nt * 16 + ln16][2 * vI + 16 * half];
#pragma unroll
    for (int mt = 0; mt < 2; ++mt)
#pragma unroll
      for (int nt = 0; nt < 2; ++nt)
        acc[mt][nt] = __builtin_amdgcn_wmma_f32_16x16x32_bf16(
            false, a[mt].v, false, b[nt].v, (short)0, acc[mt][nt], false, false);
  }

#pragma unroll
  for (int mt = 0; mt < 2; ++mt) {
    const int row  = m0 + wm * 32 + mt * 16 + 8 * half;
    const int col0 = n0 + wn * 32 + ln16;
#pragma unroll
    for (int r = 0; r < 8; ++r) {
      C[(size_t)(row + r) * N + col0]      = acc[mt][0][r];
      C[(size_t)(row + r) * N + col0 + 16] = acc[mt][1][r];
    }
  }
}

// ---------------------------------------------------------------- RMSNorm
__global__ __launch_bounds__(256) void k_rmsnorm_bf16(
    const float* __restrict__ x, const float* __restrict__ g,
    unsigned short* __restrict__ out, int D, int inStride) {
  __shared__ float red[256];
  const int t = blockIdx.x;
  const float* row = x + (size_t)t * inStride;
  float ss = 0.f;
  for (int i = threadIdx.x; i < D; i += 256) { float v = row[i]; ss += v * v; }
  red[threadIdx.x] = ss;
  __syncthreads();
  for (int s = 128; s > 0; s >>= 1) {
    if (threadIdx.x < s) red[threadIdx.x] += red[threadIdx.x + s];
    __syncthreads();
  }
  float inv = rsqrtf(red[0] / (float)D + EPS);
  for (int i = threadIdx.x; i < D; i += 256)
    out[(size_t)t * D + i] = f2bf(row[i] * inv * g[i]);
}

// ---------------------------------------------------------------- RoPE helper
__device__ __forceinline__ float rope_val(const float* __restrict__ base,
                                          int local, float pos) {
  int j = local >> 1;
  float inv = __expf(-((float)(2 * j) * (1.0f / 64.0f)) * LN10000);
  float ang = pos * inv;
  float c = __cosf(ang), s = __sinf(ang);
  float x1 = base[2 * j], x2 = base[2 * j + 1];
  return (local & 1) ? (x2 * c + x1 * s) : (x1 * c - x2 * s);
}

// ---------------------------------------------------------------- assemble q
__global__ void k_assemble_q(const float* __restrict__ q, const int* __restrict__ pos,
                             unsigned short* __restrict__ out) {
  const int t = blockIdx.x;
  const float p = (float)pos[t];
  const float* row = q + (size_t)t * (H * DQK);
  unsigned short* orow = out + (size_t)t * (H * DQK);
  for (int i = threadIdx.x; i < H * DQK; i += blockDim.x) {
    int h = i / DQK, d = i % DQK;
    float v = (d < NOPE) ? row[h * DQK + d]
                         : rope_val(row + h * DQK + NOPE, d - NOPE, p);
    orow[i] = f2bf(v);
  }
}

// ---------------------------------------------------------------- assemble k,v
__global__ void k_assemble_kv(const float* __restrict__ kv, const float* __restrict__ kva,
                              const int* __restrict__ pos,
                              unsigned short* __restrict__ kout,
                              unsigned short* __restrict__ vout) {
  const int t = blockIdx.x;
  const float p = (float)pos[t];
  const float* kvrow = kv + (size_t)t * (H * (NOPE + VD));
  const float* perow = kva + (size_t)t * (KV_RANK + ROPE) + KV_RANK;
  unsigned short* krow = kout + (size_t)t * (H * DQK);
  unsigned short* vrow = vout + (size_t)t * (H * VD);
  for (int i = threadIdx.x; i < H * DQK; i += blockDim.x) {
    int h = i / DQK, d = i % DQK;
    float v = (d < NOPE) ? kvrow[h * (NOPE + VD) + d]
                         : rope_val(perow, d - NOPE, p);
    krow[i] = f2bf(v);
  }
  for (int i = threadIdx.x; i < H * VD; i += blockDim.x) {
    int h = i / VD, d = i % VD;
    vrow[i] = f2bf(kvrow[h * (NOPE + VD) + NOPE + d]);
  }
}

// ---------------------------------------------------------------- flash attention
// block = (head, 128 q rows); 8 waves, 16 q rows per wave; keys in blocks of 32.
// Double-buffered K/V staging: async DMA of block i+1 overlaps WMMA on block i.
__global__ __launch_bounds__(256) void k_attention(
    const unsigned short* __restrict__ q,   // [T, H*192]
    const unsigned short* __restrict__ k,   // [T, H*192]
    const unsigned short* __restrict__ v,   // [T, H*128]
    unsigned short* __restrict__ o) {       // [T, H*128]
  __shared__ unsigned short Klds[2][32][DQK];  // key-major
  __shared__ unsigned short VldsT[2][VD][32];  // feature-major (transposed)
  __shared__ unsigned short Plds[8][16][32];

  const int h    = blockIdx.y;
  const int qb   = blockIdx.x * 128;
  const int tid  = threadIdx.x;
  const int wave = tid >> 5;
  const int lane = tid & 31;
  const int half = lane >> 4;
  const int ln16 = lane & 15;
  const int wq0  = qb + wave * 16;

  // Q fragments: 6 chunks of K=32 over d=192, loaded once
  Frag16 qf[6];
  {
    const unsigned short* qrow = q + (size_t)(wq0 + ln16) * (H * DQK) + h * DQK;
#pragma unroll
    for (int kc = 0; kc < 6; ++kc)
#pragma unroll
      for (int vI = 0; vI < 8; ++vI)
        qf[kc].u[vI] = *(const unsigned int*)&qrow[kc * 32 + 2 * vI + 16 * half];
  }

  v8f oacc[8] = {};
  float mrow[8], lrow[8];
#pragma unroll
  for (int r = 0; r < 8; ++r) { mrow[r] = -1e30f; lrow[r] = 0.f; }

  auto stageKV = [&](int buf, int kb) {
    // K block 32x192 via async global->LDS (768 b128 chunks, 3 per thread)
    for (int i = tid; i < 32 * 24; i += 256) {
      int rr = i / 24, cc = (i % 24) * 8;
      async_copy_b128(&k[(size_t)(kb + rr) * (H * DQK) + h * DQK + cc],
                      &Klds[buf][rr][cc]);
    }
    for (int i = tid; i < 32 * 16; i += 256) {            // V block 32x128 -> T
      int rr = i / 16, cc = (i % 16) * 8;
      uint4 tv = *(const uint4*)&v[(size_t)(kb + rr) * (H * VD) + h * VD + cc];
      const unsigned short* pv = (const unsigned short*)&tv;
#pragma unroll
      for (int j = 0; j < 8; ++j) VldsT[buf][cc + j][rr] = pv[j];
    }
  };

  stageKV(0, 0);
  int buf = 0;
  const int kend = qb + 128;
  for (int kb = 0; kb < kend; kb += 32, buf ^= 1) {
    async_wait0();
    __syncthreads();
    if (kb + 32 < kend) stageKV(buf ^ 1, kb + 32);

    if (kb <= wq0 + 15) {             // else: fully causal-masked for this wave
      // S = Q K^T : two 16x16 key tiles, 6 WMMAs each over d=192
      v8f s0 = {}, s1 = {};
#pragma unroll
      for (int kc = 0; kc < 6; ++kc) {
        Frag16 b0, b1;
#pragma unroll
        for (int vI = 0; vI < 8; ++vI) {
          b0.u[vI] = *(const unsigned int*)&Klds[buf][ln16][kc * 32 + 2 * vI + 16 * half];
          b1.u[vI] = *(const unsigned int*)&Klds[buf][16 + ln16][kc * 32 + 2 * vI + 16 * half];
        }
        s0 = __builtin_amdgcn_wmma_f32_16x16x32_bf16(false, qf[kc].v, false, b0.v,
                                                     (short)0, s0, false, false);
        s1 = __builtin_amdgcn_wmma_f32_16x16x32_bf16(false, qf[kc].v, false, b1.v,
                                                     (short)0, s1, false, false);
      }

      // scale + causal mask + online softmax (row lives across 16 lanes per half)
      float pfac[8];
#pragma unroll
      for (int r = 0; r < 8; ++r) {
        int qrow = wq0 + r + 8 * half;
        float a0 = (kb + ln16      <= qrow) ? s0[r] * SCALE : -1e30f;
        float a1 = (kb + 16 + ln16 <= qrow) ? s1[r] * SCALE : -1e30f;
        float mx = fmaxf(a0, a1);
#pragma unroll
        for (int d = 8; d >= 1; d >>= 1) mx = fmaxf(mx, __shfl_xor(mx, d, 32));
        float mnew = fmaxf(mrow[r], mx);
        float corr = __expf(mrow[r] - mnew);
        mrow[r] = mnew;
        float p0 = __expf(a0 - mnew);
        float p1 = __expf(a1 - mnew);
        s0[r] = p0; s1[r] = p1;
        float sum = p0 + p1;
#pragma unroll
        for (int d = 8; d >= 1; d >>= 1) sum += __shfl_xor(sum, d, 32);
        lrow[r] = lrow[r] * corr + sum;
        pfac[r] = corr;
      }
#pragma unroll
      for (int nt = 0; nt < 8; ++nt)
#pragma unroll
        for (int r = 0; r < 8; ++r) oacc[nt][r] *= pfac[r];

      // repack P: C layout -> A layout through per-wave LDS staging
#pragma unroll
      for (int r = 0; r < 8; ++r) {
        Plds[wave][r + 8 * half][ln16]      = f2bf(s0[r]);
        Plds[wave][r + 8 * half][16 + ln16] = f2bf(s1[r]);
      }
      __builtin_amdgcn_wave_barrier();
      Frag16 pf;
#pragma unroll
      for (int vI = 0; vI < 8; ++vI)
        pf.u[vI] = *(const unsigned int*)&Plds[wave][ln16][2 * vI + 16 * half];

      // O += P V : 8 WMMAs over d_v=128
#pragma unroll
      for (int nt = 0; nt < 8; ++nt) {
        Frag16 bv;
#pragma unroll
        for (int vI = 0; vI < 8; ++vI)
          bv.u[vI] = *(const unsigned int*)&VldsT[buf][nt * 16 + ln16][2 * vI + 16 * half];
        oacc[nt] = __builtin_amdgcn_wmma_f32_16x16x32_bf16(false, pf.v, false, bv.v,
                                                           (short)0, oacc[nt], false, false);
      }
    }
  }

#pragma unroll
  for (int r = 0; r < 8; ++r) {
    float invl = 1.0f / lrow[r];
    int trow = wq0 + r + 8 * half;
    unsigned short* orow = o + (size_t)trow * (H * VD) + h * VD;
#pragma unroll
    for (int nt = 0; nt < 8; ++nt)
      orow[nt * 16 + ln16] = f2bf(oacc[nt][r] * invl);
  }
}

// ---------------------------------------------------------------- launch
extern "C" void kernel_launch(void* const* d_in, const int* in_sizes, int n_in,
                              void* d_out, int out_size, void* d_ws, size_t ws_size,
                              hipStream_t stream) {
  const float* hidden    = (const float*)d_in[0];
  const float* w_qa      = (const float*)d_in[1];
  const float* g_qa      = (const float*)d_in[2];
  const float* w_qb      = (const float*)d_in[3];
  const float* w_kva     = (const float*)d_in[4];
  const float* g_kva     = (const float*)d_in[5];
  const float* w_kvb     = (const float*)d_in[6];
  const float* w_o       = (const float*)d_in[7];
  const int*   positions = (const int*)d_in[8];

  char* p = (char*)d_ws;
  auto alloc = [&](size_t bytes) -> char* {
    char* r = p; p += (bytes + 255) & ~(size_t)255; return r;
  };
  unsigned short* hid_bf  = (unsigned short*)alloc((size_t)T_TOK * HID * 2);
  unsigned short* wqa_bf  = (unsigned short*)alloc((size_t)HID * Q_RANK * 2);
  unsigned short* wqb_bf  = (unsigned short*)alloc((size_t)Q_RANK * H * DQK * 2);
  unsigned short* wkva_bf = (unsigned short*)alloc((size_t)HID * (KV_RANK + ROPE) * 2);
  unsigned short* wkvb_bf = (unsigned short*)alloc((size_t)KV_RANK * H * (NOPE + VD) * 2);
  unsigned short* wo_bf   = (unsigned short*)alloc((size_t)H * VD * HID * 2);
  float*          qa_f    = (float*)alloc((size_t)T_TOK * Q_RANK * 4);
  unsigned short* qan_bf  = (unsigned short*)alloc((size_t)T_TOK * Q_RANK * 2);
  float*          q_f     = (float*)alloc((size_t)T_TOK * H * DQK * 4);
  unsigned short* q_bf    = (unsigned short*)alloc((size_t)T_TOK * H * DQK * 2);
  float*          kva_f   = (float*)alloc((size_t)T_TOK * (KV_RANK + ROPE) * 4);
  unsigned short* kvan_bf = (unsigned short*)alloc((size_t)T_TOK * KV_RANK * 2);
  float*          kv_f    = (float*)alloc((size_t)T_TOK * H * (NOPE + VD) * 4);
  unsigned short* k_bf    = (unsigned short*)alloc((size_t)T_TOK * H * DQK * 2);
  unsigned short* v_bf    = (unsigned short*)alloc((size_t)T_TOK * H * VD * 2);
  unsigned short* o_bf    = (unsigned short*)alloc((size_t)T_TOK * H * VD * 2);
  (void)ws_size; (void)n_in; (void)in_sizes; (void)out_size;

  auto cvt = [&](const float* src, unsigned short* dst, size_t n) {
    k_cvt_bf16<<<1024, 256, 0, stream>>>(src, dst, (int)n);
  };
  cvt(hidden, hid_bf,  (size_t)T_TOK * HID);
  cvt(w_qa,   wqa_bf,  (size_t)HID * Q_RANK);
  cvt(w_qb,   wqb_bf,  (size_t)Q_RANK * H * DQK);
  cvt(w_kva,  wkva_bf, (size_t)HID * (KV_RANK + ROPE));
  cvt(w_kvb,  wkvb_bf, (size_t)KV_RANK * H * (NOPE + VD));
  cvt(w_o,    wo_bf,   (size_t)H * VD * HID);

  auto gemm = [&](const unsigned short* A, const unsigned short* B, float* C,
                  int M, int N, int K) {
    k_gemm_bf16<<<dim3(N / 64, M / 128), 256, 0, stream>>>(A, B, C, M, N, K);
  };

  // Q path
  gemm(hid_bf, wqa_bf, qa_f, T_TOK, Q_RANK, HID);
  k_rmsnorm_bf16<<<T_TOK, 256, 0, stream>>>(qa_f, g_qa, qan_bf, Q_RANK, Q_RANK);
  gemm(qan_bf, wqb_bf, q_f, T_TOK, H * DQK, Q_RANK);
  // KV path
  gemm(hid_bf, wkva_bf, kva_f, T_TOK, KV_RANK + ROPE, HID);
  k_rmsnorm_bf16<<<T_TOK, 256, 0, stream>>>(kva_f, g_kva, kvan_bf, KV_RANK, KV_RANK + ROPE);
  gemm(kvan_bf, wkvb_bf, kv_f, T_TOK, H * (NOPE + VD), KV_RANK);
  // RoPE + assembly
  k_assemble_q<<<T_TOK, 256, 0, stream>>>(q_f, positions, q_bf);
  k_assemble_kv<<<T_TOK, 256, 0, stream>>>(kv_f, kva_f, positions, k_bf, v_bf);
  // Flash attention
  k_attention<<<dim3(T_TOK / 128, H), 256, 0, stream>>>(q_bf, k_bf, v_bf, o_bf);
  // Output projection -> f32 d_out
  gemm(o_bf, wo_bf, (float*)d_out, T_TOK, HID, H * VD);
}
